// HolographicMixer_63934883168960
// MI455X (gfx1250) — matrix operand order repaired and checked by reading, not compiled, not run on hardware
//
// HolographicMixer for MI455X (gfx1250, wave32, WMMA).
//
// out = x + Re(ifft( Fs ⊙ (conj(Fq) − |Fr|²) )),  Fs = Σ_s fft(K)⊙fft(V)
// (exact algebraic reduction of the reference; Hermitian symmetry of real
// signals makes every ifft exactly real, and fft(Re(ifft(H))) = H.)
//
// Pass 1: fused bf16-WMMA GEMM (K,V) + tanh + packed in-place FFT(K+iV) +
//         Fk⊙Fv, deterministic per-block partials -> reduce kernel -> Fs.
// Pass 2: fused bf16-WMMA GEMM (Q,R) + tanh + packed FFT(Q+iR) + spectral
//         combine with Fs + in-place DIF inverse FFT + residual add -> out.
//
// 137 GFLOP of GEMM dominates vs ~200 MB HBM traffic (9 us at 23.3 TB/s):
// compute-bound -> bf16 WMMA. This round: the GEMM epilogue scatters tanh
// values bit-reversed straight into the packed z = K+iV stage rows (one
// float2[1024] per row), the FFT runs in place there, and the inverse FFT is
// DIF (natural in / bit-reversed out). That removes all separate FFT buffers:
// LDS drops to 136 KB (pass 1) / 144 KB (pass 2) -> 2 blocks per WGP ->
// 4 waves/SIMD of latency hiding behind the loadcnt waits seen in round-3 asm.

#include <hip/hip_runtime.h>

typedef __attribute__((ext_vector_type(16))) __bf16 v16bf;
typedef __attribute__((ext_vector_type(8)))  __bf16 v8bf;
typedef __attribute__((ext_vector_type(4)))  __bf16 v4bf;
typedef __attribute__((ext_vector_type(8)))  float  v8f;

static __device__ __forceinline__ v8f wmma_bf16(v16bf a, v16bf b, v8f c) {
  // args: (neg_a, A, neg_b, B, c_mod, C, reuse_a, reuse_b)
  return __builtin_amdgcn_wmma_f32_16x16x32_bf16(false, a, false, b, (short)0, c,
                                                 false, false);
}

// A-matrix 16x32 bf16 fragment, contiguous k order (matches B):
// lanes 0-15: M=lane, K[0..15]
// lanes16-31: M=lane-16, K[16..31]
// => per lane: one contiguous 32B run at k = 16*half.
static __device__ __forceinline__ v16bf load_a_frag(const __bf16* p) {
  v8bf lo = *(const v8bf*)(p);
  v8bf hi = *(const v8bf*)(p + 8);
  v16bf a;
#pragma unroll
  for (int i = 0; i < 8; ++i) { a[i] = lo[i]; a[i + 8] = hi[i]; }
  return a;
}

__global__ __launch_bounds__(256) void cvt_f32_bf16_kernel(
    const float* __restrict__ src, __bf16* __restrict__ dst, int n) {
  int i = (blockIdx.x * blockDim.x + threadIdx.x) * 4;
  if (i + 3 < n) {
    float4 f = *(const float4*)(src + i);
    v4bf o = { (__bf16)f.x, (__bf16)f.y, (__bf16)f.z, (__bf16)f.w };
    *(v4bf*)(dst + i) = o;
  }
}

// Shared twiddle table: tw[h + j] = exp(-i*pi*j/h), h = 1,2,...,512.
static __device__ __forceinline__ void build_twiddles(float2* tw, int tid) {
  for (int idx = tid; idx < 1024; idx += 256) {
    float2 t;
    if (idx == 0) {
      t = make_float2(1.0f, 0.0f);
    } else {
      int h = 1 << (31 - __clz(idx));
      int j = idx - h;
      float ang = -3.14159265358979323846f * (float)j / (float)h;
      float sv, cv;
      __sincosf(ang, &sv, &cv);
      t = make_float2(cv, sv);
    }
    tw[idx] = t;
  }
}

// In-place radix-2 DIT forward FFT (1024 pts) by one wave32.
// Input bit-reverse permuted (done by the GEMM epilogue scatter); output
// natural order. Same-wave LDS ops complete in order on CDNA5; s_wait_dscnt
// + "memory" clobber pins the compiler between stages.
static __device__ __forceinline__ void fft1024_dit_wave(float2* ws,
                                                        const float2* tw,
                                                        int lane) {
#pragma unroll 1
  for (int s = 1; s <= 10; ++s) {
    const int halfl = 1 << (s - 1);
#pragma unroll 1
    for (int idx = lane; idx < 512; idx += 32) {
      int j  = idx & (halfl - 1);
      int i0 = ((idx >> (s - 1)) << s) + j;
      int i1 = i0 + halfl;
      float2 w = tw[halfl + j];
      float2 u = ws[i0];
      float2 v = ws[i1];
      float tr = w.x * v.x - w.y * v.y;
      float ti = w.x * v.y + w.y * v.x;
      ws[i0] = make_float2(u.x + tr, u.y + ti);
      ws[i1] = make_float2(u.x - tr, u.y - ti);
    }
    asm volatile("s_wait_dscnt 0x0" ::: "memory");
  }
}

// In-place radix-2 DIF inverse FFT (unscaled): natural-order input,
// bit-reversed output, conjugate twiddles. Caller applies 1/N and the
// bit-reversed read at the final store.
static __device__ __forceinline__ void ifft1024_dif_wave(float2* ws,
                                                         const float2* tw,
                                                         int lane) {
#pragma unroll 1
  for (int s = 10; s >= 1; --s) {
    const int halfl = 1 << (s - 1);
#pragma unroll 1
    for (int idx = lane; idx < 512; idx += 32) {
      int j  = idx & (halfl - 1);
      int i0 = ((idx >> (s - 1)) << s) + j;
      int i1 = i0 + halfl;
      float2 w = tw[halfl + j];            // use conj(w) => inverse
      float2 u = ws[i0];
      float2 v = ws[i1];
      float dr = u.x - v.x;
      float di = u.y - v.y;
      ws[i0] = make_float2(u.x + v.x, u.y + v.y);
      ws[i1] = make_float2(w.x * dr + w.y * di, w.x * di - w.y * dr);
    }
    asm volatile("s_wait_dscnt 0x0" ::: "memory");
  }
}

// Software-pipelined dual-matrix GEMM inner loop (shared by both passes).
// acc0/acc1 accumulate A*B0 and A*B1 for 8 N-tiles each.
// b0base/b1base point at this wave's lane row: W + (w*128 + l16)*1024 + 16*half.
// Lookahead-2: the fragment pair for logical tile g is loaded at step g-2 and
// consumed at step g. The tail lookahead overreads <=~36KB into the adjacent
// workspace region (mapped, results unused) to keep the hot loop branch-free.
static __device__ __forceinline__ void gemm_dual_pipelined(
    const __bf16* __restrict__ xA, const __bf16* __restrict__ b0base,
    const __bf16* __restrict__ b1base, v8f acc0[8], v8f acc1[8]) {
  v16bf a_cur = load_a_frag(xA);
  // Pipeline preload: logical tiles 0 and 1 of k0 = 0.
  v16bf f0_0 = *(const v16bf*)(b0base);
  v16bf f1_0 = *(const v16bf*)(b1base);
  v16bf f0_1 = *(const v16bf*)(b0base + 16 * 1024);
  v16bf f1_1 = *(const v16bf*)(b1base + 16 * 1024);
#pragma unroll 1
  for (int k0 = 0; k0 < 1024; k0 += 32) {
    v16bf a_next = load_a_frag(xA + k0 + 32);          // next k-step A
    __builtin_prefetch(xA + k0 + 64, 0, 3);            // near-scope prefetch
#pragma unroll
    for (int t = 0; t < 8; ++t) {
      // Issue loads for logical tile t+2 (wraps into the next k-step); t is
      // compile-time constant after unroll, so no branches are emitted.
      const int idx = t + 2;
      const size_t noff = (idx < 8)
                              ? ((size_t)idx * 16 * 1024 + k0)
                              : ((size_t)(idx - 8) * 16 * 1024 + (k0 + 32));
      v16bf n0 = *(const v16bf*)(b0base + noff);
      v16bf n1 = *(const v16bf*)(b1base + noff);
      acc0[t] = wmma_bf16(a_cur, f0_0, acc0[t]);
      acc1[t] = wmma_bf16(a_cur, f1_0, acc1[t]);
      f0_0 = f0_1;  f0_1 = n0;                         // SSA shift (renamed)
      f1_0 = f1_1;  f1_1 = n1;
    }
    a_cur = a_next;
  }
}

// ---------------- Pass 1: K,V GEMM + FFT + Fk*Fv partial sums ---------------
// Block: 256 threads (8 waves), tile = 16 rows x 1024 cols for BOTH K and V.
// Wave w owns N columns [w*128, w*128+128) -> 8 N-tiles x 2 matrices,
// 16 x v8f accumulators (128 VGPRs/lane). LDS = 136 KB -> 2 blocks/WGP.
__global__ __launch_bounds__(256) void gemm_kv_fft_kernel(
    const __bf16* __restrict__ xb,
    const __bf16* __restrict__ wkb, const __bf16* __restrict__ wvb,
    const float* __restrict__ biask, const float* __restrict__ biasv,
    float* __restrict__ partial) {
  extern __shared__ char smem[];
  float2* stageZ = (float2*)smem;                // 16 rows * 1024 c64 (128 KB)
  float2* tw     = stageZ + 16 * 1024;           // 8 KB

  const int tid  = threadIdx.x;
  const int w    = tid >> 5;
  const int lane = tid & 31;
  const int half = lane >> 4;
  const int l16  = lane & 15;
  const int row0 = blockIdx.x * 16;              // 16384 rows / 16 = 1024 blocks

  build_twiddles(tw, tid);

  v8f ak[8], av[8];
#pragma unroll
  for (int t = 0; t < 8; ++t) {
    ak[t] = (v8f)0.0f;
    av[t] = (v8f)0.0f;
  }

  // A rows are shared by all 8 waves; read direct from global (WGP$-resident).
  const __bf16* xA = xb + (size_t)(row0 + l16) * 1024 + 16 * half;
  // B-matrix 32x16: lanes0-15 N=lane K[0..15], lanes16-31 N=lane-16 K[16..31]
  // -> one contiguous 32B load per lane from a W row (W stored [N,K]).
  const __bf16* bkB = wkb + (size_t)(w * 128 + l16) * 1024 + 16 * half;
  const __bf16* bvB = wvb + (size_t)(w * 128 + l16) * 1024 + 16 * half;
  gemm_dual_pipelined(xA, bkB, bvB, ak, av);

  // Epilogue: bias + tanh, scatter packed z = K + iV (one ds_store_b64 per
  // element) at the BIT-REVERSED column so the in-place DIT FFT needs no
  // separate permutation buffer. C frag: M = 8*half + r, N = l16.
#pragma unroll
  for (int t = 0; t < 8; ++t) {
    int col = w * 128 + t * 16 + l16;
    int rc  = __brev(col) >> 22;                 // 10-bit bit reversal
    float bk = biask[col], bv = biasv[col];
#pragma unroll
    for (int r = 0; r < 8; ++r) {
      int m = 8 * half + r;
      stageZ[m * 1024 + rc] = make_float2(tanhf(ak[t][r] + bk),
                                          tanhf(av[t][r] + bv));
    }
  }
  __syncthreads();

  // In-place FFT + Hermitian pairwise product; each wave owns rows {w, w+8}.
#pragma unroll 1
  for (int rr = 0; rr < 2; ++rr) {
    float2* zrow = stageZ + (w + rr * 8) * 1024;
    fft1024_dit_wave(zrow, tw, lane);
    // Pairs (i, 1024-i) are each owned by exactly one lane -> in-place safe.
    // P[1024-i] = conj(P[i]) exactly (keeps summed Fs exactly Hermitian).
#pragma unroll 1
    for (int i = lane; i <= 512; i += 32) {
      int mi = (1024 - i) & 1023;
      float2 zi = zrow[i], zm = zrow[mi];
      float fkre = 0.5f * (zi.x + zm.x);
      float fkim = 0.5f * (zi.y - zm.y);
      float fvre = 0.5f * (zi.y + zm.y);
      float fvim = -0.5f * (zi.x - zm.x);
      float pre = fkre * fvre - fkim * fvim;
      float pim = fkre * fvim + fkim * fvre;
      zrow[i]  = make_float2(pre, pim);
      zrow[mi] = make_float2(pre, -pim);         // self-pairs: pim == 0
    }
  }
  __syncthreads();

  // Deterministic in-block reduction over the 16 rows -> per-block partial.
  float* pout = partial + (size_t)blockIdx.x * 2048;
#pragma unroll 1
  for (int i = tid; i < 1024; i += 256) {
    float sre = 0.0f, sim = 0.0f;
#pragma unroll
    for (int r = 0; r < 16; ++r) {
      float2 p = stageZ[r * 1024 + i];
      sre += p.x;
      sim += p.y;
    }
    pout[2 * i]     = sre;
    pout[2 * i + 1] = sim;
  }
}

// Deterministic cross-block reduction: Fs[b] = sum of 256 block partials.
__global__ __launch_bounds__(256) void reduce_fs_kernel(
    const float* __restrict__ partial, float* __restrict__ Fs) {
  int i = blockIdx.x * 256 + threadIdx.x;        // 0..2047
  int b = blockIdx.y;                            // batch
  const float* p = partial + (size_t)b * 256 * 2048 + i;
  float s = 0.0f;
#pragma unroll 4
  for (int blk = 0; blk < 256; ++blk) s += p[(size_t)blk * 2048];
  Fs[(size_t)b * 2048 + i] = s;
}

// ------------- Pass 2: Q,R GEMM + FFT + spectral combine + iFFT -------------
// LDS = 144 KB -> 2 blocks/WGP.
__global__ __launch_bounds__(256) void gemm_qr_out_kernel(
    const __bf16* __restrict__ xb,
    const __bf16* __restrict__ wqb, const __bf16* __restrict__ wrb,
    const float* __restrict__ biasq, const float* __restrict__ biasr,
    const float* __restrict__ xf, const float* __restrict__ Fs,
    float* __restrict__ out) {
  extern __shared__ char smem[];
  float2* stageZ = (float2*)smem;                // 128 KB
  float2* tw     = stageZ + 16 * 1024;           // 8 KB
  float*  fsl    = (float*)(tw + 1024);          // 8 KB (Fs for this batch)

  const int tid  = threadIdx.x;
  const int w    = tid >> 5;
  const int lane = tid & 31;
  const int half = lane >> 4;
  const int l16  = lane & 15;
  const int row0 = blockIdx.x * 16;
  const int batch = row0 >> 12;                  // 4096 rows per batch

  build_twiddles(tw, tid);

  // Fs table (8 KB) -> LDS via the CDNA5 async-to-LDS path (ASYNCcnt).
  // Dynamic LDS starts at offset 0 (no static __shared__), so the LDS address
  // operand is just the byte offset of fsl within the dynamic segment.
  {
    const float* fsg = Fs + (size_t)batch * 2048;
    unsigned base_off = (unsigned)((const char*)fsl - (const char*)smem);
    unsigned long long gaddr = (unsigned long long)fsg;
    for (int i = tid; i < 512; i += 256) {       // 512 x 16B = 8 KB
      unsigned lds_a = base_off + (unsigned)i * 16u;
      unsigned voff  = (unsigned)i * 16u;
      asm volatile("global_load_async_to_lds_b128 %0, %1, %2 offset:0"
                   :: "v"(lds_a), "v"(voff), "s"(gaddr)
                   : "memory");
    }
    asm volatile("s_wait_asynccnt 0x0" ::: "memory");
  }

  v8f aq[8], ar[8];
#pragma unroll
  for (int t = 0; t < 8; ++t) {
    aq[t] = (v8f)0.0f;
    ar[t] = (v8f)0.0f;
  }

  const __bf16* xA  = xb + (size_t)(row0 + l16) * 1024 + 16 * half;
  const __bf16* bqB = wqb + (size_t)(w * 128 + l16) * 1024 + 16 * half;
  const __bf16* brB = wrb + (size_t)(w * 128 + l16) * 1024 + 16 * half;
  gemm_dual_pipelined(xA, bqB, brB, aq, ar);

  // Epilogue: packed z = Q + iR scattered bit-reversed.
#pragma unroll
  for (int t = 0; t < 8; ++t) {
    int col = w * 128 + t * 16 + l16;
    int rc  = __brev(col) >> 22;
    float bq = biasq[col], br = biasr[col];
#pragma unroll
    for (int r = 0; r < 8; ++r) {
      int m = 8 * half + r;
      stageZ[m * 1024 + rc] = make_float2(tanhf(aq[t][r] + bq),
                                          tanhf(ar[t][r] + br));
    }
  }
  __syncthreads();                               // also publishes fsl

  const float invN = 1.0f / 1024.0f;
#pragma unroll 1
  for (int rr = 0; rr < 2; ++rr) {
    const int r = w + rr * 8;
    float2* zrow = stageZ + r * 1024;
    const size_t grow = (size_t)(row0 + r) * 1024;
    fft1024_dit_wave(zrow, tw, lane);
    // g[i] = Fs[i] * (conj(Fq[i]) - |Fr[i]|^2) in natural order, in place;
    // g[1024-i] = conj(g[i]) exactly (Fs is exactly Hermitian by pass 1).
#pragma unroll 1
    for (int i = lane; i <= 512; i += 32) {
      int mi = (1024 - i) & 1023;
      float2 zi = zrow[i], zm = zrow[mi];
      float fqre = 0.5f * (zi.x + zm.x);
      float fqim = 0.5f * (zi.y - zm.y);
      float frre = 0.5f * (zi.y + zm.y);
      float frim = -0.5f * (zi.x - zm.x);
      float cre = fqre - (frre * frre + frim * frim);
      float cim = -fqim;
      float fsre = fsl[2 * i], fsim = fsl[2 * i + 1];
      float gre = fsre * cre - fsim * cim;
      float gim = fsre * cim + fsim * cre;
      zrow[i]  = make_float2(gre, gim);
      zrow[mi] = make_float2(gre, -gim);         // self-pairs: gim == 0
    }
    asm volatile("s_wait_dscnt 0x0" ::: "memory");
    ifft1024_dif_wave(zrow, tw, lane);           // natural in, bit-rev out
    // out = x + Re(ifft(g)) / N  (exactly real by Hermitian symmetry);
    // read LDS at the bit-reversed position, store natural order.
#pragma unroll 1
    for (int i = lane; i < 1024; i += 32) {
      int src = __brev(i) >> 22;
      out[grow + i] = xf[grow + i] + zrow[src].x * invN;
    }
  }
}

// ------------------------------- launcher -----------------------------------
extern "C" void kernel_launch(void* const* d_in, const int* in_sizes, int n_in,
                              void* d_out, int out_size, void* d_ws, size_t ws_size,
                              hipStream_t stream) {
  (void)in_sizes; (void)n_in; (void)out_size; (void)ws_size;
  const float* x  = (const float*)d_in[0];
  const float* Wq = (const float*)d_in[1];
  const float* bq = (const float*)d_in[2];
  const float* Wk = (const float*)d_in[3];
  const float* bk = (const float*)d_in[4];
  const float* Wv = (const float*)d_in[5];
  const float* bv = (const float*)d_in[6];
  const float* Wr = (const float*)d_in[7];
  const float* br = (const float*)d_in[8];
  float* out = (float*)d_out;

  // Workspace layout (needs ~50.4 MB):
  //   xb      : 16M bf16            (32 MB)
  //   wq/k/v/r: 4 x 1M bf16         ( 8 MB)
  //   partial : 1024 blk x 2048 f32 ( 8 MB)
  //   Fs      : 4 x 2048 f32        (32 KB)
  char* wsb = (char*)d_ws;
  __bf16* xb  = (__bf16*)wsb;
  __bf16* wqb = (__bf16*)(wsb + 33554432);
  __bf16* wkb = wqb + 1048576;
  __bf16* wvb = wkb + 1048576;
  __bf16* wrb = wvb + 1048576;
  float* partial = (float*)(wsb + 33554432 + 8388608);
  float* Fs      = partial + (size_t)1024 * 2048;

  // bf16 conversion (4 elems/thread)
  cvt_f32_bf16_kernel<<<16384, 256, 0, stream>>>(x,  xb,  16777216);
  cvt_f32_bf16_kernel<<<1024,  256, 0, stream>>>(Wq, wqb, 1048576);
  cvt_f32_bf16_kernel<<<1024,  256, 0, stream>>>(Wk, wkb, 1048576);
  cvt_f32_bf16_kernel<<<1024,  256, 0, stream>>>(Wv, wvb, 1048576);
  cvt_f32_bf16_kernel<<<1024,  256, 0, stream>>>(Wr, wrb, 1048576);

  // Pass 1: LDS = 128K z-stage + 8K twiddle = 139264 B (2 blocks/WGP)
  gemm_kv_fft_kernel<<<1024, 256, 139264, stream>>>(xb, wkb, wvb, bk, bv, partial);
  reduce_fs_kernel<<<dim3(8, 4), 256, 0, stream>>>(partial, Fs);

  // Pass 2: LDS = 128K z-stage + 8K tw + 8K Fs = 147456 B (2 blocks/WGP)
  gemm_qr_out_kernel<<<1024, 256, 147456, stream>>>(xb, wqb, wrb, bq, br, x, Fs, out);
}
